// DiscreteAutoregressiveFlow_59957743452790
// MI455X (gfx1250) — matrix operand order repaired and checked by reading, not compiled, not run on hardware
//
#include <hip/hip_runtime.h>

typedef __attribute__((ext_vector_type(16))) _Float16 v16h;
typedef __attribute__((ext_vector_type(8)))  float    v8f;

#define B_SZ  1024
#define L_SZ  1024
#define V_SZ  20
#define D_SZ  128
#define NOUT  40   // 2*V

#define NEG_INF (-__builtin_inff())

// ---------------------------------------------------------------------------
// Pass 1: one-hot [B,L,V] -> token index uint16 [B,L]  (first-max tie rule)
// ---------------------------------------------------------------------------
__global__ void daf_tokenize(const float* __restrict__ x,
                             unsigned short* __restrict__ tok) {
  int gid = blockIdx.x * blockDim.x + threadIdx.x;           // gid = b*L + l
  const float* p = x + (size_t)gid * V_SZ;                   // 80B, 16B aligned
  float v[V_SZ];
#pragma unroll
  for (int q = 0; q < 5; ++q) {
    float4 f = *(const float4*)(p + 4 * q);
    v[4 * q + 0] = f.x; v[4 * q + 1] = f.y;
    v[4 * q + 2] = f.z; v[4 * q + 3] = f.w;
  }
  float bv = v[0]; int bi = 0;
#pragma unroll
  for (int j = 1; j < V_SZ; ++j)
    if (v[j] > bv) { bv = v[j]; bi = j; }                    // strict > = first max
  tok[gid] = (unsigned short)bi;
}

// ---------------------------------------------------------------------------
// Pass 2: sequential scan. One wave32 per 16 batch rows.
//   net^T[48x16] = Wout^T[48x128] @ state^T[128x16] + bias
//   A (Wout^T) loop-invariant in 96 VGPRs (contiguous-k f16 A-layout).
//   state^T = B operand: lane (n=lane&15, hi=lane>>4) owns
//   state[n][32c + 16*hi + h] as f32 shadow, repacked to v16h per step.
// ---------------------------------------------------------------------------
__launch_bounds__(32)
__global__ void daf_scan(const float* __restrict__ emb,
                         const float* __restrict__ Wout,
                         const float* __restrict__ bvec,
                         const unsigned short* __restrict__ tok,
                         float* __restrict__ out) {
  __shared__ float ldsEmb[V_SZ * D_SZ];   // 10 KB
  __shared__ int   ldsInv[V_SZ];

  const int lane = threadIdx.x;
  const int lo   = lane & 15;
  const int hi   = lane >> 4;

  // ---- stage emb + mult-inverse table into LDS (once) ----
  for (int i = lane; i < V_SZ * D_SZ; i += 32) ldsEmb[i] = emb[i];
  if (lane < V_SZ) {
    int inv = 0;
    for (int m = 1; m < V_SZ; ++m) if ((lane * m) % V_SZ == 1) inv = m;
    ldsInv[lane] = inv;                    // 0 when not coprime (matches ref)
  }
  __syncthreads();

  // ---- A operand: Wout^T, 3 M-tiles x 4 K-chunks, contiguous-k 16-bit A layout ----
  // A[m][k] = Wout[k*40 + m]; element h <-> K = 32*ch + 16*hi + h
  v16h A[3][4];
#pragma unroll
  for (int mt = 0; mt < 3; ++mt) {
    const int m = mt * 16 + lo;
#pragma unroll
    for (int ch = 0; ch < 4; ++ch) {
      v16h a;
#pragma unroll
      for (int h = 0; h < 16; ++h) {
        const int k = 32 * ch + 16 * hi + h;
        const float w = (m < NOUT) ? Wout[k * NOUT + m] : 0.0f;
        a[h] = (_Float16)w;
      }
      A[mt][ch] = a;
    }
  }

  // ---- bias in C layout: VGPR r holds row mt*16 + 8*hi + r ----
  float bias[3][8];
#pragma unroll
  for (int mt = 0; mt < 3; ++mt)
#pragma unroll
    for (int r = 0; r < 8; ++r) {
      const int row = mt * 16 + 8 * hi + r;
      bias[mt][r] = (row < NOUT) ? bvec[row] : 0.0f;
    }

  // ---- loop-carried state: f32 shadow + packed f16 B operands ----
  float s[64];
#pragma unroll
  for (int i = 0; i < 64; ++i) s[i] = 0.0f;
  v16h Bst[4];
#pragma unroll
  for (int ch = 0; ch < 4; ++ch)
#pragma unroll
    for (int h = 0; h < 16; ++h) Bst[ch][h] = (_Float16)0.0f;

  const int brow = blockIdx.x * 16 + lo;
  const unsigned short* trow = tok + (size_t)brow * L_SZ;

  int a_tok = (int)trow[0];                // software-pipelined token stream

  for (int t = 0; t < L_SZ; ++t) {
    const int a_next = (t + 1 < L_SZ) ? (int)trow[t + 1] : 0;  // prefetch t+1

    // ---- net^T = bias + Wout^T @ state^T : 12 x v_wmma_f32_16x16x32_f16 ----
    v8f c0, c1, c2;
#pragma unroll
    for (int r = 0; r < 8; ++r) { c0[r] = bias[0][r]; c1[r] = bias[1][r]; c2[r] = bias[2][r]; }
#pragma unroll
    for (int ch = 0; ch < 4; ++ch) {
      c0 = __builtin_amdgcn_wmma_f32_16x16x32_f16(false, A[0][ch], false, Bst[ch], (short)0, c0, false, false);
      c1 = __builtin_amdgcn_wmma_f32_16x16x32_f16(false, A[1][ch], false, Bst[ch], (short)0, c1, false, false);
      c2 = __builtin_amdgcn_wmma_f32_16x16x32_f16(false, A[2][ch], false, Bst[ch], (short)0, c2, false, false);
    }

    // ---- argmax(loc), branchless. net col j: c0[r] -> j = 8*hi + r;
    //      c1[r] (r<4, hi=0 only) -> j = 16 + r. Per-lane ascending order. ----
    float lv = NEG_INF; int li = 0;
#pragma unroll
    for (int r = 0; r < 8; ++r) {
      const float cv = c0[r];
      const int   ci = 8 * hi + r;
      if (cv > lv) { lv = cv; li = ci; }
    }
#pragma unroll
    for (int r = 0; r < 4; ++r) {
      const float cv = hi ? NEG_INF : c1[r];
      if (cv > lv) { lv = cv; li = 16 + r; }
    }
    {
      const float pv = __shfl_xor(lv, 16, 32);
      const int   pi = __shfl_xor(li, 16, 32);
      if (pv > lv || (pv == lv && pi < li)) { lv = pv; li = pi; }
    }
    const int loc = li;

    // ---- argmax(scale), branchless. s-index: c1[r] -> s = hi ? r+4 : r-4
    //      (r<4 invalid when hi=0); c2[r] (hi=0 only) -> s = 12 + r. ----
    float sv = NEG_INF; int si = 0;
#pragma unroll
    for (int r = 0; r < 8; ++r) {
      const float cv = (r < 4) ? (hi ? c1[r] : NEG_INF) : c1[r];
      const int   ci = hi ? (r + 4) : (r - 4);
      if (cv > sv) { sv = cv; si = ci; }
    }
#pragma unroll
    for (int r = 0; r < 8; ++r) {
      const float cv = hi ? NEG_INF : c2[r];
      if (cv > sv) { sv = cv; si = 12 + r; }
    }
    {
      const float pv = __shfl_xor(sv, 16, 32);
      const int   pi = __shfl_xor(si, 16, 32);
      if (pv > sv || (pv == sv && pi < si)) { sv = pv; si = pi; }
    }
    const int sidx = si;

    // ---- modular one-hot algebra: u = inv(s) * ((a - loc) mod V) mod V ----
    int mm = a_tok - loc + V_SZ; if (mm >= V_SZ) mm -= V_SZ;
    const int u = (ldsInv[sidx] * mm) % V_SZ;

    // ---- write full one-hot row, branch-free: every lane 2x b128 + 1x b64.
    //      hi=0: v0..3 @+0, v4..7 @+16, v8..9 @+32
    //      hi=1: v12..15 @+48, v16..19 @+64, v10..11 @+40 ----
    {
      float* orow = out + ((size_t)brow * L_SZ + t) * V_SZ;
      const int v4a = 12 * hi;          // first float4 base index
      const int v4b = 4 + 12 * hi;      // second float4 base index
      const int v2c = 8 + 2 * hi;       // float2 base index
      float4 fa = make_float4(v4a + 0 == u ? 1.0f : 0.0f,
                              v4a + 1 == u ? 1.0f : 0.0f,
                              v4a + 2 == u ? 1.0f : 0.0f,
                              v4a + 3 == u ? 1.0f : 0.0f);
      float4 fb = make_float4(v4b + 0 == u ? 1.0f : 0.0f,
                              v4b + 1 == u ? 1.0f : 0.0f,
                              v4b + 2 == u ? 1.0f : 0.0f,
                              v4b + 3 == u ? 1.0f : 0.0f);
      float2 fc = make_float2(v2c + 0 == u ? 1.0f : 0.0f,
                              v2c + 1 == u ? 1.0f : 0.0f);
      *(float4*)(orow + v4a) = fa;      // +0B  or +48B (16B aligned)
      *(float4*)(orow + v4b) = fb;      // +16B or +64B (16B aligned)
      *(float2*)(orow + v2c) = fc;      // +32B or +40B (8B aligned)
    }

    // ---- state += emb[u]; repack f16 B operands for next step ----
#pragma unroll
    for (int ch = 0; ch < 4; ++ch) {
      const float* e = &ldsEmb[u * D_SZ + 32 * ch + 16 * hi];
#pragma unroll
      for (int q = 0; q < 4; ++q) {
        float4 ev = *(const float4*)(e + 4 * q);
        s[ch * 16 + 4 * q + 0] += ev.x;
        s[ch * 16 + 4 * q + 1] += ev.y;
        s[ch * 16 + 4 * q + 2] += ev.z;
        s[ch * 16 + 4 * q + 3] += ev.w;
      }
      v16h nb;
#pragma unroll
      for (int h = 0; h < 16; ++h) nb[h] = (_Float16)s[ch * 16 + h];
      Bst[ch] = nb;
    }

    a_tok = a_next;
  }
}

// ---------------------------------------------------------------------------
extern "C" void kernel_launch(void* const* d_in, const int* in_sizes, int n_in,
                              void* d_out, int out_size, void* d_ws, size_t ws_size,
                              hipStream_t stream) {
  const float* x    = (const float*)d_in[0];   // [B,L,V]
  const float* emb  = (const float*)d_in[1];   // [V,D]
  const float* Wout = (const float*)d_in[2];   // [D,2V]
  const float* bvec = (const float*)d_in[3];   // [2V]
  float* out = (float*)d_out;                  // [B,L,V]
  unsigned short* tok = (unsigned short*)d_ws; // [B,L] scratch (2 MB)

  (void)in_sizes; (void)n_in; (void)out_size; (void)ws_size;

  const int total = B_SZ * L_SZ;
  daf_tokenize<<<total / 256, 256, 0, stream>>>(x, tok);
  daf_scan<<<B_SZ / 16, 32, 0, stream>>>(emb, Wout, bvec, tok, out);
}
